// GAT3NamesEdgesCentralityLSTM_38465727103219
// MI455X (gfx1250) — matrix operand order repaired and checked by reading, not compiled, not run on hardware
//
#include <hip/hip_runtime.h>
#include <hip/hip_bf16.h>
#include <math.h>

// ---------------------------------------------------------------------------
// GAT3 + GlobalAttentionPooling + LSTM head for MI455X (gfx1250, wave32).
//
// Roofline: edge aggregation (E=800K x 256 f32 gather+atomic per layer) is the
// bottleneck; ft/rst (51MB each) are L2-resident (192MB L2). The node GEMMs
// ([N,64]x[64,256]) use v_wmma_f32_16x16x32_bf16 with LDS operands staged
// directly in WMMA fragment layout (lane-contiguous 32B -> ds_load_b128).
// Two col-tiles are processed per iteration with independent accumulators to
// hide the WMMA->WMMA accumulation hazard.
// ---------------------------------------------------------------------------

typedef __attribute__((ext_vector_type(16))) __bf16 v16bf;
typedef __attribute__((ext_vector_type(8)))  float  v8f;

#define HIDD   64
#define NHEAD  4
#define HOUT   256      // NHEAD*HIDD
#define BGR    64       // graphs in batch
#define ENC_NEG_INF 0x007FFFFFu   // fenc(-inf)

__device__ __forceinline__ unsigned fenc(float f) {
  unsigned u = __float_as_uint(f);
  return (u & 0x80000000u) ? ~u : (u | 0x80000000u);
}
__device__ __forceinline__ float fdec(unsigned u) {
  u = (u & 0x80000000u) ? (u & 0x7fffffffu) : ~u;
  return __uint_as_float(u);
}
__device__ __forceinline__ float sigm(float x) { return 1.f / (1.f + __expf(-x)); }

// -------------------------------- utility ---------------------------------
__global__ void k_fill_u32(unsigned* __restrict__ p, long n, unsigned v) {
  long i = blockIdx.x * (long)blockDim.x + threadIdx.x;
  if (i < n) p[i] = v;
}

// x = concat(nfeats[N,48], names[N,16]) * centrality[N,1]
__global__ void k_concat(const float* __restrict__ nf, const float* __restrict__ nm,
                         const float* __restrict__ ce, float* __restrict__ x, int n) {
  int t = blockIdx.x * blockDim.x + threadIdx.x;
  if (t >= n * HIDD) return;
  int node = t >> 6, d = t & 63;
  float v = (d < 48) ? nf[node * 48 + d] : nm[node * 16 + (d - 48)];
  x[t] = v * ce[node];
}

// ------------------------- WMMA GEMM: FT = X @ W ---------------------------
// X:[n,64] f32, W:[64,256] f32, FT:[n,256] f32.
// Block = 256 threads (8 waves) = 128 rows. W staged once per block into LDS
// pre-swizzled into B-fragment layout; X rows staged into A-fragment layout.
// Each wave owns one 16-row M-tile and sweeps all 16 col-tiles, two tiles per
// iteration with independent accumulators (hazard hiding): 32 v_wmma / wave.
//
// Fragment layouts as staged here (contiguous K, not the ISA table order; 16-bit, wave32):
//   A 16x32: lane = half*16 + m (m=row), elem i -> K = half*16 + i
//   B 32x16: lane = half*16 + m (m=col), elem i -> K = half*16 + i
//   C 16x16: lane = half*16 + m (m=col), elem j -> row = half*8 + j
__global__ void __launch_bounds__(256)
k_gemm_ft(const float* __restrict__ X, const float* __restrict__ W,
          float* __restrict__ FT, int n) {
  // WB[ct][kfrag][lane][i] : 16*2*32*16 bf16 = 32 KB
  __shared__ __align__(32) __bf16 WB[16 * 2 * 32 * 16];
  // XA[wave][kfrag][lane][i] : 8*2*32*16 bf16 = 16 KB
  __shared__ __align__(32) __bf16 XA[8 * 2 * 32 * 16];

  const int tid  = threadIdx.x;
  const int row0 = blockIdx.x * 128;

  // stage W (64x256) into B-fragment layout
  for (int idx = tid; idx < 64 * HOUT; idx += 256) {
    int k = idx >> 8, col = idx & 255;
    int ct = col >> 4, m = col & 15;
    int kfrag = k >> 5, kr = k & 31;
    int half = kr >> 4, i = kr & 15;
    WB[((((ct << 1) + kfrag) << 5) + (half << 4) + m) * 16 + i] = (__bf16)W[idx];
  }
  // stage X rows [row0, row0+128) into A-fragment layout
  for (int idx = tid; idx < 128 * HIDD; idx += 256) {
    int r = idx >> 6, k = idx & 63;
    int row = row0 + r;
    float v = (row < n) ? X[(long)row * HIDD + k] : 0.f;
    int w = r >> 4, rl = r & 15;
    int kfrag = k >> 5, kr = k & 31;
    int half = kr >> 4, i = kr & 15;
    XA[((((w << 1) + kfrag) << 5) + (half << 4) + rl) * 16 + i] = (__bf16)v;
  }
  __syncthreads();

  const int wave = tid >> 5, lane = tid & 31;
  const int half = lane >> 4, m = lane & 15;

  const v16bf a0 = *(const v16bf*)&XA[((((wave << 1) + 0) << 5) + lane) * 16];
  const v16bf a1 = *(const v16bf*)&XA[((((wave << 1) + 1) << 5) + lane) * 16];

  const int  wrow0 = row0 + wave * 16 + half * 8;     // first row this lane stores
  const bool full  = (row0 + wave * 16 + 16) <= n;    // uniform per wave

  if (full) {
#pragma unroll
    for (int ct = 0; ct < 16; ct += 2) {
      const v16bf b0a = *(const v16bf*)&WB[((((ct << 1) + 0) << 5) + lane) * 16];
      const v16bf b1a = *(const v16bf*)&WB[((((ct << 1) + 1) << 5) + lane) * 16];
      const v16bf b0b = *(const v16bf*)&WB[((((ct << 1) + 2) << 5) + lane) * 16];
      const v16bf b1b = *(const v16bf*)&WB[((((ct << 1) + 3) << 5) + lane) * 16];
      v8f c0 = {}, c1 = {};
      c0 = __builtin_amdgcn_wmma_f32_16x16x32_bf16(false, a0, false, b0a, (short)0, c0, false, false);
      c1 = __builtin_amdgcn_wmma_f32_16x16x32_bf16(false, a0, false, b0b, (short)0, c1, false, false);
      c0 = __builtin_amdgcn_wmma_f32_16x16x32_bf16(false, a1, false, b1a, (short)0, c0, false, false);
      c1 = __builtin_amdgcn_wmma_f32_16x16x32_bf16(false, a1, false, b1b, (short)0, c1, false, false);
      float* outp = FT + (long)wrow0 * HOUT + (ct << 4) + m;
#pragma unroll
      for (int j = 0; j < 8; ++j) outp[j * HOUT] = c0[j];
#pragma unroll
      for (int j = 0; j < 8; ++j) outp[j * HOUT + 16] = c1[j];
    }
  } else {
#pragma unroll
    for (int ct = 0; ct < 16; ct += 2) {
      const v16bf b0a = *(const v16bf*)&WB[((((ct << 1) + 0) << 5) + lane) * 16];
      const v16bf b1a = *(const v16bf*)&WB[((((ct << 1) + 1) << 5) + lane) * 16];
      const v16bf b0b = *(const v16bf*)&WB[((((ct << 1) + 2) << 5) + lane) * 16];
      const v16bf b1b = *(const v16bf*)&WB[((((ct << 1) + 3) << 5) + lane) * 16];
      v8f c0 = {}, c1 = {};
      c0 = __builtin_amdgcn_wmma_f32_16x16x32_bf16(false, a0, false, b0a, (short)0, c0, false, false);
      c1 = __builtin_amdgcn_wmma_f32_16x16x32_bf16(false, a0, false, b0b, (short)0, c1, false, false);
      c0 = __builtin_amdgcn_wmma_f32_16x16x32_bf16(false, a1, false, b1a, (short)0, c0, false, false);
      c1 = __builtin_amdgcn_wmma_f32_16x16x32_bf16(false, a1, false, b1b, (short)0, c1, false, false);
      float* outp = FT + (long)wrow0 * HOUT + (ct << 4) + m;
#pragma unroll
      for (int j = 0; j < 8; ++j)
        if (wrow0 + j < n) outp[j * HOUT] = c0[j];
#pragma unroll
      for (int j = 0; j < 8; ++j)
        if (wrow0 + j < n) outp[j * HOUT + 16] = c1[j];
    }
  }
}

// el[n,h] = ft[n,h,:].al[h,:]   er likewise
__global__ void k_attn(const float* __restrict__ FT, const float* __restrict__ AL,
                       const float* __restrict__ AR, float* __restrict__ EL,
                       float* __restrict__ ER, int n) {
  int t = blockIdx.x * blockDim.x + threadIdx.x;
  if (t >= n * NHEAD) return;
  int node = t >> 2, h = t & 3;
  const float* f  = FT + (long)node * HOUT + h * HIDD;
  const float* al = AL + h * HIDD;
  const float* ar = AR + h * HIDD;
  float sl = 0.f, sr = 0.f;
#pragma unroll 4
  for (int d = 0; d < HIDD; ++d) { float v = f[d]; sl += v * al[d]; sr += v * ar[d]; }
  EL[t] = sl; ER[t] = sr;
}

// pass 1: e = leakyrelu(el[src]+er[dst]); segment max over dst
__global__ void k_edge_logits(const int* __restrict__ src, const int* __restrict__ dst,
                              const float* __restrict__ EL, const float* __restrict__ ER,
                              float* __restrict__ EV, unsigned* __restrict__ MU, int e) {
  int t = blockIdx.x * blockDim.x + threadIdx.x;
  if (t >= e) return;
  int s = src[t], d = dst[t];
#pragma unroll
  for (int h = 0; h < NHEAD; ++h) {
    float v = EL[s * 4 + h] + ER[d * 4 + h];
    v = v > 0.f ? v : 0.2f * v;
    EV[t * 4 + h] = v;
    atomicMax(&MU[d * 4 + h], fenc(v));
  }
}

// pass 2: ex = exp(e - max); segment sum over dst
__global__ void k_edge_exp(const int* __restrict__ dst, float* __restrict__ EV,
                           const unsigned* __restrict__ MU, float* __restrict__ S, int e) {
  int t = blockIdx.x * blockDim.x + threadIdx.x;
  if (t >= e) return;
  int d = dst[t];
#pragma unroll
  for (int h = 0; h < NHEAD; ++h) {
    float ex = __expf(EV[t * 4 + h] - fdec(MU[d * 4 + h]));
    EV[t * 4 + h] = ex;
    atomicAdd(&S[d * 4 + h], ex);
  }
}

// pass 3: rst[dst] += (softmax * ew) * ft[src];  one warp per edge, 8 cols/lane
__global__ void __launch_bounds__(256)
k_edge_agg(const int* __restrict__ src, const int* __restrict__ dst,
           const float* __restrict__ ew, const float* __restrict__ EV,
           const float* __restrict__ S, const float* __restrict__ FT,
           float* __restrict__ RST, int e) {
  int edge = blockIdx.x * 8 + (threadIdx.x >> 5);
  if (edge >= e) return;
  int lane = threadIdx.x & 31;
  int s = src[edge], d = dst[edge];
  float w = ew[edge];
  float a[NHEAD];
#pragma unroll
  for (int h = 0; h < NHEAD; ++h) a[h] = EV[edge * 4 + h] / S[d * 4 + h] * w;
#pragma unroll
  for (int j = 0; j < 8; ++j) {
    int c = lane + j * 32;
    atomicAdd(&RST[(long)d * HOUT + c], a[c >> 6] * FT[(long)s * HOUT + c]);
  }
}

// h[n,d] = mean_h relu(rst[n,h,d] + b[h,d])
__global__ void k_finalize(const float* __restrict__ RST, const float* __restrict__ bias,
                           float* __restrict__ out, int n) {
  int t = blockIdx.x * blockDim.x + threadIdx.x;
  if (t >= n * HIDD) return;
  int node = t >> 6, d = t & 63;
  float acc = 0.f;
#pragma unroll
  for (int h = 0; h < NHEAD; ++h) {
    float v = RST[(long)node * HOUT + h * HIDD + d] + bias[h * HIDD + d];
    acc += v > 0.f ? v : 0.f;
  }
  out[t] = acc * 0.25f;
}

// ---------------------- GlobalAttentionPooling (3 passes) ------------------
__global__ void k_gate_logit(const float* __restrict__ x, const float* __restrict__ Wg,
                             const float* __restrict__ bg, const int* __restrict__ gid,
                             float* __restrict__ GL, unsigned* __restrict__ MG, int n) {
  int t = blockIdx.x * blockDim.x + threadIdx.x;
  if (t >= n) return;
  const float* xp = x + (long)t * HIDD;
  float acc = bg[0];
#pragma unroll 4
  for (int d = 0; d < HIDD; ++d) acc += xp[d] * Wg[d];
  GL[t] = acc;
  atomicMax(&MG[gid[t]], fenc(acc));
}

__global__ void k_gate_exp(const int* __restrict__ gid, float* __restrict__ GL,
                           const unsigned* __restrict__ MG, float* __restrict__ SG, int n) {
  int t = blockIdx.x * blockDim.x + threadIdx.x;
  if (t >= n) return;
  int g = gid[t];
  float ex = __expf(GL[t] - fdec(MG[g]));
  GL[t] = ex;
  atomicAdd(&SG[g], ex);
}

__global__ void k_gate_pool(const float* __restrict__ x, const int* __restrict__ gid,
                            const float* __restrict__ GL, const float* __restrict__ SG,
                            float* __restrict__ HG, int n) {
  int t = blockIdx.x * blockDim.x + threadIdx.x;
  if (t >= n * HIDD) return;
  int node = t >> 6, d = t & 63;
  int g = gid[node];
  atomicAdd(&HG[g * HIDD + d], (GL[node] / SG[g]) * x[t]);
}

// ------------------------------- LSTM head ---------------------------------
// one block, 64 threads (thread = batch row); h-state in padded LDS.
__global__ void __launch_bounds__(64)
k_lstm_head(const float* __restrict__ HG, const float* __restrict__ Wih,
            const float* __restrict__ Whh, const float* __restrict__ bih,
            const float* __restrict__ bhh, const float* __restrict__ Wc,
            const float* __restrict__ bc, float* __restrict__ out) {
  __shared__ float hs[BGR][HIDD + 1];
  int b = threadIdx.x;
  float cst[HIDD];
  for (int d = 0; d < HIDD; ++d) { cst[d] = 0.f; hs[b][d] = 0.f; }
  __syncthreads();
  for (int t = 0; t < 4; ++t) {
    const float* x = HG + (long)t * BGR * HIDD + (long)b * HIDD;
    float nh[HIDD];
    for (int d = 0; d < HIDD; ++d) {
      float gi = bih[d]       + bhh[d];
      float gf = bih[64 + d]  + bhh[64 + d];
      float gg = bih[128 + d] + bhh[128 + d];
      float go = bih[192 + d] + bhh[192 + d];
      for (int k = 0; k < HIDD; ++k) {
        float xv = x[k], hv = hs[b][k];
        gi += xv * Wih[d * 64 + k]         + hv * Whh[d * 64 + k];
        gf += xv * Wih[(64 + d) * 64 + k]  + hv * Whh[(64 + d) * 64 + k];
        gg += xv * Wih[(128 + d) * 64 + k] + hv * Whh[(128 + d) * 64 + k];
        go += xv * Wih[(192 + d) * 64 + k] + hv * Whh[(192 + d) * 64 + k];
      }
      cst[d] = sigm(gf) * cst[d] + sigm(gi) * tanhf(gg);
      nh[d]  = sigm(go) * tanhf(cst[d]);
    }
    __syncthreads();
    for (int d = 0; d < HIDD; ++d) hs[b][d] = nh[d];
    __syncthreads();
  }
  float acc = bc[0];
  for (int d = 0; d < HIDD; ++d) acc += hs[b][d] * Wc[d];
  out[b] = sigm(acc);
}

// ------------------------------- launcher ----------------------------------
static inline int cdiv(long a, int b) { return (int)((a + b - 1) / b); }

extern "C" void kernel_launch(void* const* d_in, const int* in_sizes, int n_in,
                              void* d_out, int out_size, void* d_ws, size_t ws_size,
                              hipStream_t stream) {
  const int N = in_sizes[0] / 48;   // nfeats [N,48]
  const int E = in_sizes[3];        // edge_weight [E]

  const float* nfeats = (const float*)d_in[0];
  const float* names  = (const float*)d_in[1];
  const float* centr  = (const float*)d_in[2];
  const float* ew     = (const float*)d_in[3];
  const int*   src    = (const int*)d_in[4];
  const int*   dst    = (const int*)d_in[5];
  const int*   gid    = (const int*)d_in[6];
  const float* Wl[3]  = {(const float*)d_in[7],  (const float*)d_in[11], (const float*)d_in[15]};
  const float* ALl[3] = {(const float*)d_in[8],  (const float*)d_in[12], (const float*)d_in[16]};
  const float* ARl[3] = {(const float*)d_in[9],  (const float*)d_in[13], (const float*)d_in[17]};
  const float* Bl[3]  = {(const float*)d_in[10], (const float*)d_in[14], (const float*)d_in[18]};
  const float* Wg[4]  = {(const float*)d_in[19], (const float*)d_in[21], (const float*)d_in[23], (const float*)d_in[25]};
  const float* bg[4]  = {(const float*)d_in[20], (const float*)d_in[22], (const float*)d_in[24], (const float*)d_in[26]};
  const float* Wih = (const float*)d_in[27];
  const float* Whh = (const float*)d_in[28];
  const float* bih = (const float*)d_in[29];
  const float* bhh = (const float*)d_in[30];
  const float* Wc  = (const float*)d_in[31];
  const float* bc  = (const float*)d_in[32];

  // workspace layout
  char* ws = (char*)d_ws;
  size_t off = 0;
  auto take = [&](size_t bytes) -> void* {
    void* p = ws + off;
    off = (off + bytes + 255) & ~(size_t)255;
    return p;
  };
  float*    CUR = (float*)take((size_t)N * HIDD * 4);
  float*    FT  = (float*)take((size_t)N * HOUT * 4);
  float*    RST = (float*)take((size_t)N * HOUT * 4);
  float*    EL  = (float*)take((size_t)N * NHEAD * 4);
  float*    ER  = (float*)take((size_t)N * NHEAD * 4);
  float*    EV  = (float*)take((size_t)E * NHEAD * 4);
  unsigned* MU  = (unsigned*)take((size_t)N * NHEAD * 4);
  float*    S   = (float*)take((size_t)N * NHEAD * 4);
  float*    GL  = (float*)take((size_t)N * 4);
  unsigned* MG  = (unsigned*)take((size_t)BGR * 4);
  float*    SG  = (float*)take((size_t)BGR * 4);
  float*    HG  = (float*)take((size_t)4 * BGR * HIDD * 4);
  (void)ws_size; (void)n_in; (void)out_size;

  const int T = 256;
  auto gap = [&](const float* x, int l) {
    k_fill_u32<<<1, BGR, 0, stream>>>(MG, BGR, ENC_NEG_INF);
    k_fill_u32<<<1, BGR, 0, stream>>>((unsigned*)SG, BGR, 0u);
    k_fill_u32<<<cdiv(BGR * HIDD, T), T, 0, stream>>>((unsigned*)(HG + (size_t)l * BGR * HIDD),
                                                      BGR * HIDD, 0u);
    k_gate_logit<<<cdiv(N, T), T, 0, stream>>>(x, Wg[l], bg[l], gid, GL, MG, N);
    k_gate_exp<<<cdiv(N, T), T, 0, stream>>>(gid, GL, MG, SG, N);
    k_gate_pool<<<cdiv((long)N * HIDD, T), T, 0, stream>>>(x, gid, GL, SG,
                                                           HG + (size_t)l * BGR * HIDD, N);
  };

  // x = concat(nfeats,names)*centrality ; hg0
  k_concat<<<cdiv((long)N * HIDD, T), T, 0, stream>>>(nfeats, names, centr, CUR, N);
  gap(CUR, 0);

  for (int l = 0; l < 3; ++l) {
    k_gemm_ft<<<cdiv(N, 128), 256, 0, stream>>>(CUR, Wl[l], FT, N);
    k_attn<<<cdiv((long)N * NHEAD, T), T, 0, stream>>>(FT, ALl[l], ARl[l], EL, ER, N);
    k_fill_u32<<<cdiv((long)N * NHEAD, T), T, 0, stream>>>(MU, (long)N * NHEAD, ENC_NEG_INF);
    k_fill_u32<<<cdiv((long)N * NHEAD, T), T, 0, stream>>>((unsigned*)S, (long)N * NHEAD, 0u);
    k_fill_u32<<<cdiv((long)N * HOUT, T), T, 0, stream>>>((unsigned*)RST, (long)N * HOUT, 0u);
    k_edge_logits<<<cdiv(E, T), T, 0, stream>>>(src, dst, EL, ER, EV, MU, E);
    k_edge_exp<<<cdiv(E, T), T, 0, stream>>>(dst, EV, MU, S, E);
    k_edge_agg<<<cdiv(E, 8), 256, 0, stream>>>(src, dst, ew, EV, S, FT, RST, E);
    k_finalize<<<cdiv((long)N * HIDD, T), T, 0, stream>>>(RST, Bl[l], CUR, N);
    gap(CUR, l + 1);
  }

  k_lstm_head<<<1, BGR, 0, stream>>>(HG, Wih, Whh, bih, bhh, Wc, bc, (float*)d_out);
}
